// FAGCN_41729902247983
// MI455X (gfx1250) — matrix operand order repaired and checked by reading, not compiled, not run on hardware
//
#include <hip/hip_runtime.h>
#include <hip/hip_bf16.h>
#include <math.h>

// ---------------------------------------------------------------------------
// FAGCN forward for MI455X (gfx1250, wave32).
//  - GEMMs via V_WMMA_F32_16X16X32_BF16 (native v_cvt bf16 converts, f32 accum).
//  - Edge scatter via per-wave gather + native global_atomic_add_f32.
//  - Exact stable top-k prune per column via rank counting.
// ---------------------------------------------------------------------------

#define NNODE 32768
#define NEDGE 262144
#define NFEAT 512
#define NHID  256
#define NCLS  40
#define VLEN  1024
#define WLEN  32
#define EPS_F 0.1f

typedef __attribute__((ext_vector_type(16))) __bf16 v16bf;
typedef __attribute__((ext_vector_type(8)))  float  v8f;

// Native f32 -> bf16 (RTNE) — lowers to v_cvt_*bf16_f32 on gfx1250.
__device__ __forceinline__ __bf16 f2bf(float f) { return (__bf16)f; }

__device__ __forceinline__ float wave_reduce_sum(float v) {
#pragma unroll
  for (int off = 16; off > 0; off >>= 1)
    v += __shfl_xor(v, off, 32);
  return v;
}

// -------------------------------------------------------------------- init --
__global__ void init_masks_kernel(float* __restrict__ topidx,
                                  float* __restrict__ emask) {
  const int i = blockIdx.x * blockDim.x + threadIdx.x;
  if (i < NEDGE) emask[i] = 1.0f;
  if (i < NNODE) topidx[i] = 1.0f;
}

// ------------------------------------------------- GEMM1: h = relu(xW^T+b) --
// One wave computes a 16(M) x 64(N) strip; 4 accumulators share the A frag.
__global__ void __launch_bounds__(256)
gemm_start_kernel(const float* __restrict__ x, const float* __restrict__ Wst,
                  const float* __restrict__ bst, float* __restrict__ h,
                  float* __restrict__ h0) {
  const int wave = (blockIdx.x * blockDim.x + threadIdx.x) >> 5;
  const int lane = threadIdx.x & 31;
  const int half = lane >> 4;
  const int li   = lane & 15;
  const int mt = wave >> 2;        // 0..2047
  const int nt = wave & 3;         // 0..3
  const int m0 = mt << 4;
  const int n0 = nt << 6;
  const float* xrow = x + (size_t)(m0 + li) * NFEAT;

  v8f acc[4] = {{}, {}, {}, {}};

  for (int kk = 0; kk < NFEAT; kk += 32) {
    __builtin_prefetch(xrow + kk + 32, 0, 0);  // global_prefetch_b8
    const int ka = kk + half * 16;
    float4 fa0 = *(const float4*)(xrow + ka);
    float4 fa1 = *(const float4*)(xrow + ka + 4);
    float4 fa2 = *(const float4*)(xrow + ka + 8);
    float4 fa3 = *(const float4*)(xrow + ka + 12);
    v16bf a;
    a[0]=f2bf(fa0.x); a[1]=f2bf(fa0.y); a[2]=f2bf(fa0.z); a[3]=f2bf(fa0.w);
    a[4]=f2bf(fa1.x); a[5]=f2bf(fa1.y); a[6]=f2bf(fa1.z); a[7]=f2bf(fa1.w);
    a[8]=f2bf(fa2.x); a[9]=f2bf(fa2.y); a[10]=f2bf(fa2.z); a[11]=f2bf(fa2.w);
    a[12]=f2bf(fa3.x); a[13]=f2bf(fa3.y); a[14]=f2bf(fa3.z); a[15]=f2bf(fa3.w);

    const int k0 = kk + half * 16;
#pragma unroll
    for (int j = 0; j < 4; ++j) {
      const int n = n0 + (j << 4) + li;
      const float* wrow = Wst + (size_t)n * NFEAT + k0;  // B[k][n] = W[n][k]
      float4 fb0 = *(const float4*)(wrow);
      float4 fb1 = *(const float4*)(wrow + 4);
      float4 fb2 = *(const float4*)(wrow + 8);
      float4 fb3 = *(const float4*)(wrow + 12);
      v16bf b;
      b[0]=f2bf(fb0.x); b[1]=f2bf(fb0.y); b[2]=f2bf(fb0.z); b[3]=f2bf(fb0.w);
      b[4]=f2bf(fb1.x); b[5]=f2bf(fb1.y); b[6]=f2bf(fb1.z); b[7]=f2bf(fb1.w);
      b[8]=f2bf(fb2.x); b[9]=f2bf(fb2.y); b[10]=f2bf(fb2.z); b[11]=f2bf(fb2.w);
      b[12]=f2bf(fb3.x); b[13]=f2bf(fb3.y); b[14]=f2bf(fb3.z); b[15]=f2bf(fb3.w);
      acc[j] = __builtin_amdgcn_wmma_f32_16x16x32_bf16(
          false, a, false, b, (short)0, acc[j], false, false);
    }
  }
#pragma unroll
  for (int j = 0; j < 4; ++j) {
    const int col = n0 + (j << 4) + li;
    const float bias = bst[col];
#pragma unroll
    for (int v = 0; v < 8; ++v) {
      const int row = m0 + half * 8 + v;   // C/D layout: M = half*8 + vgpr
      float val = acc[j][v] + bias;
      val = val > 0.f ? val : 0.f;
      h [(size_t)row * NHID + col] = val;
      h0[(size_t)row * NHID + col] = val;
    }
  }
}

// ------------------------------------------ al = h.att_l ; ar = h.att_r ----
__global__ void __launch_bounds__(256)
att_dot_kernel(const float* __restrict__ h, const float* __restrict__ attl,
               const float* __restrict__ attr_, float* __restrict__ al,
               float* __restrict__ ar) {
  const int node = (blockIdx.x * blockDim.x + threadIdx.x) >> 5;
  const int lane = threadIdx.x & 31;
  const float* hrow = h + (size_t)node * NHID + lane * 8;
  float4 a0 = *(const float4*)hrow;
  float4 a1 = *(const float4*)(hrow + 4);
  float4 l0 = *(const float4*)(attl + lane * 8);
  float4 l1 = *(const float4*)(attl + lane * 8 + 4);
  float4 r0 = *(const float4*)(attr_ + lane * 8);
  float4 r1 = *(const float4*)(attr_ + lane * 8 + 4);
  float sl = a0.x*l0.x + a0.y*l0.y + a0.z*l0.z + a0.w*l0.w
           + a1.x*l1.x + a1.y*l1.y + a1.z*l1.z + a1.w*l1.w;
  float sr = a0.x*r0.x + a0.y*r0.y + a0.z*r0.z + a0.w*r0.w
           + a1.x*r1.x + a1.y*r1.y + a1.z*r1.z + a1.w*r1.w;
  sl = wave_reduce_sum(sl);
  sr = wave_reduce_sum(sr);
  if (lane == 0) { al[node] = sl; ar[node] = sr; }
}

// --------------- agg[dst] += h[src] * tanh(al[src]+ar[dst])*w per edge -----
__global__ void __launch_bounds__(256)
edge_scatter_kernel(const int* __restrict__ eidx, const float* __restrict__ eattr,
                    const float* __restrict__ emask, const float* __restrict__ al,
                    const float* __restrict__ ar, const float* __restrict__ h,
                    float* __restrict__ agg) {
  const int e    = (blockIdx.x * blockDim.x + threadIdx.x) >> 5;
  const int lane = threadIdx.x & 31;
  const float m = emask[e];
  if (m == 0.f) return;                       // pruned edge: whole wave exits
  const int s = eidx[e];
  const int d = eidx[NEDGE + e];
  const float coef = tanhf(al[s] + ar[d]) * eattr[e] * m;
  if (coef == 0.f) return;
  const float* hrow = h + (size_t)s * NHID + lane * 8;
  float4 v0 = *(const float4*)hrow;
  float4 v1 = *(const float4*)(hrow + 4);
  float* arow = agg + (size_t)d * NHID + lane * 8;
  unsafeAtomicAdd(arow + 0, v0.x * coef);     // global_atomic_add_f32
  unsafeAtomicAdd(arow + 1, v0.y * coef);
  unsafeAtomicAdd(arow + 2, v0.z * coef);
  unsafeAtomicAdd(arow + 3, v0.w * coef);
  unsafeAtomicAdd(arow + 4, v1.x * coef);
  unsafeAtomicAdd(arow + 5, v1.y * coef);
  unsafeAtomicAdd(arow + 6, v1.z * coef);
  unsafeAtomicAdd(arow + 7, v1.w * coef);
}

// --------- h = (agg + EPS*h0) * top_idx ; norms[i] = ||h[i]|| --------------
__global__ void __launch_bounds__(256)
combine_norm_kernel(const float* __restrict__ agg, const float* __restrict__ h0,
                    const float* __restrict__ topidx, float* __restrict__ h,
                    float* __restrict__ norms) {
  const int node = (blockIdx.x * blockDim.x + threadIdx.x) >> 5;
  const int lane = threadIdx.x & 31;
  const float t = topidx[node];
  const size_t base = (size_t)node * NHID + lane * 8;
  float4 g0 = *(const float4*)(agg + base);
  float4 g1 = *(const float4*)(agg + base + 4);
  float4 z0 = *(const float4*)(h0 + base);
  float4 z1 = *(const float4*)(h0 + base + 4);
  float4 o0, o1;
  o0.x = (g0.x + EPS_F * z0.x) * t;  o0.y = (g0.y + EPS_F * z0.y) * t;
  o0.z = (g0.z + EPS_F * z0.z) * t;  o0.w = (g0.w + EPS_F * z0.w) * t;
  o1.x = (g1.x + EPS_F * z1.x) * t;  o1.y = (g1.y + EPS_F * z1.y) * t;
  o1.z = (g1.z + EPS_F * z1.z) * t;  o1.w = (g1.w + EPS_F * z1.w) * t;
  *(float4*)(h + base)     = o0;
  *(float4*)(h + base + 4) = o1;
  float ss = o0.x*o0.x + o0.y*o0.y + o0.z*o0.z + o0.w*o0.w
           + o1.x*o1.x + o1.y*o1.y + o1.z*o1.z + o1.w*o1.w;
  ss = wave_reduce_sum(ss);
  if (lane == 0) norms[node] = sqrtf(ss);
}

// --------- per-column keep-top-k (stable argsort semantics) ----------------
__global__ void __launch_bounds__(1024)
select_topk_kernel(const float* __restrict__ norms, float* __restrict__ topidx,
                   int keep) {
  __shared__ float vals[VLEN];
  const int col = blockIdx.x;          // 0..31
  const int t   = threadIdx.x;         // 0..1023 (row within column)
  const float my = norms[t * WLEN + col];
  vals[t] = my;
  __syncthreads();
  int rank = 0;
  for (int u = 0; u < VLEN; ++u) {
    const float vu = vals[u];
    rank += (vu > my) || (vu == my && u < t);   // stable tie-break on index
  }
  if (rank >= keep) topidx[t * WLEN + col] = 0.f;
}

// --------- h *= top_idx (rows) ---------------------------------------------
__global__ void apply_mask_kernel(float* __restrict__ h,
                                  const float* __restrict__ topidx) {
  const int idx = blockIdx.x * blockDim.x + threadIdx.x;  // NHID == 256
  h[idx] *= topidx[idx >> 8];
}

// --------- edge_mask *= top_idx[src]*top_idx[dst] --------------------------
__global__ void edge_mask_kernel(const int* __restrict__ eidx,
                                 const float* __restrict__ topidx,
                                 float* __restrict__ emask) {
  const int e = blockIdx.x * blockDim.x + threadIdx.x;
  if (e < NEDGE) emask[e] *= topidx[eidx[e]] * topidx[eidx[NEDGE + e]];
}

// --------- GEMM2: out = (h W_end^T + b_end) * top_idx ----------------------
// One wave per 16-row tile; 3 n-subtiles cover NCLS=40 (padded to 48).
__global__ void __launch_bounds__(256)
gemm_end_kernel(const float* __restrict__ h, const float* __restrict__ Wend,
                const float* __restrict__ bend, const float* __restrict__ topidx,
                float* __restrict__ out) {
  const int wave = (blockIdx.x * blockDim.x + threadIdx.x) >> 5;  // m-tile
  const int lane = threadIdx.x & 31;
  const int half = lane >> 4;
  const int li   = lane & 15;
  const int m0 = wave << 4;
  const float* hrow = h + (size_t)(m0 + li) * NHID;

  v8f acc[3] = {{}, {}, {}};

  for (int kk = 0; kk < NHID; kk += 32) {
    const int ka = kk + half * 16;
    float4 fa0 = *(const float4*)(hrow + ka);
    float4 fa1 = *(const float4*)(hrow + ka + 4);
    float4 fa2 = *(const float4*)(hrow + ka + 8);
    float4 fa3 = *(const float4*)(hrow + ka + 12);
    v16bf a;
    a[0]=f2bf(fa0.x); a[1]=f2bf(fa0.y); a[2]=f2bf(fa0.z); a[3]=f2bf(fa0.w);
    a[4]=f2bf(fa1.x); a[5]=f2bf(fa1.y); a[6]=f2bf(fa1.z); a[7]=f2bf(fa1.w);
    a[8]=f2bf(fa2.x); a[9]=f2bf(fa2.y); a[10]=f2bf(fa2.z); a[11]=f2bf(fa2.w);
    a[12]=f2bf(fa3.x); a[13]=f2bf(fa3.y); a[14]=f2bf(fa3.z); a[15]=f2bf(fa3.w);

    const int k0 = kk + half * 16;
#pragma unroll
    for (int j = 0; j < 3; ++j) {
      const int n = (j << 4) + li;
      v16bf b;
      if (n < NCLS) {
        const float* wrow = Wend + (size_t)n * NHID + k0;
        float4 fb0 = *(const float4*)(wrow);
        float4 fb1 = *(const float4*)(wrow + 4);
        float4 fb2 = *(const float4*)(wrow + 8);
        float4 fb3 = *(const float4*)(wrow + 12);
        b[0]=f2bf(fb0.x); b[1]=f2bf(fb0.y); b[2]=f2bf(fb0.z); b[3]=f2bf(fb0.w);
        b[4]=f2bf(fb1.x); b[5]=f2bf(fb1.y); b[6]=f2bf(fb1.z); b[7]=f2bf(fb1.w);
        b[8]=f2bf(fb2.x); b[9]=f2bf(fb2.y); b[10]=f2bf(fb2.z); b[11]=f2bf(fb2.w);
        b[12]=f2bf(fb3.x); b[13]=f2bf(fb3.y); b[14]=f2bf(fb3.z); b[15]=f2bf(fb3.w);
      } else {
#pragma unroll
        for (int q = 0; q < 16; ++q) b[q] = (__bf16)0.0f;
      }
      // reconverged: WMMA executes with full EXEC
      acc[j] = __builtin_amdgcn_wmma_f32_16x16x32_bf16(
          false, a, false, b, (short)0, acc[j], false, false);
    }
  }
#pragma unroll
  for (int j = 0; j < 3; ++j) {
    const int col = (j << 4) + li;
    if (col < NCLS) {
      const float bias = bend[col];
#pragma unroll
      for (int v = 0; v < 8; ++v) {
        const int row = m0 + half * 8 + v;
        out[(size_t)row * NCLS + col] = (acc[j][v] + bias) * topidx[row];
      }
    }
  }
}

// ---------------------------------------------------------------------------
extern "C" void kernel_launch(void* const* d_in, const int* in_sizes, int n_in,
                              void* d_out, int out_size, void* d_ws, size_t ws_size,
                              hipStream_t stream) {
  const float* x     = (const float*)d_in[0];
  const int*   eidx  = (const int*)  d_in[1];   // (2, E) int32
  const float* eattr = (const float*)d_in[2];
  const float* Wst   = (const float*)d_in[3];
  const float* bst   = (const float*)d_in[4];
  const float* attl  = (const float*)d_in[5];   // (2, NHID)
  const float* attr_ = (const float*)d_in[6];
  const float* Wend  = (const float*)d_in[7];
  const float* bend  = (const float*)d_in[8];
  float* out = (float*)d_out;

  float* ws = (float*)d_ws;
  const size_t HN = (size_t)NNODE * NHID;       // 8,388,608 floats
  float* h      = ws;
  float* h0     = h + HN;
  float* agg    = h0 + HN;
  float* al     = agg + HN;
  float* ar     = al + NNODE;
  float* topidx = ar + NNODE;
  float* norms  = topidx + NNODE;
  float* emask  = norms + NNODE;                // NEDGE floats

  init_masks_kernel<<<NEDGE / 256, 256, 0, stream>>>(topidx, emask);
  gemm_start_kernel<<<(NNODE / 16) * 4 / 8, 256, 0, stream>>>(x, Wst, bst, h, h0);

  for (int l = 0; l < 2; ++l) {
    hipMemsetAsync(agg, 0, HN * sizeof(float), stream);
    att_dot_kernel<<<NNODE / 8, 256, 0, stream>>>(h, attl + l * NHID,
                                                  attr_ + l * NHID, al, ar);
    edge_scatter_kernel<<<NEDGE / 8, 256, 0, stream>>>(eidx, eattr, emask,
                                                       al, ar, h, agg);
    combine_norm_kernel<<<NNODE / 8, 256, 0, stream>>>(agg, h0, topidx, h, norms);
    const int keep = (l == 0) ? 256 : 128;      // ceil(1024*0.25/(l+1))
    select_topk_kernel<<<WLEN, VLEN, 0, stream>>>(norms, topidx, keep);
    apply_mask_kernel<<<(NNODE * NHID) / 256, 256, 0, stream>>>(h, topidx);
    if (l == 0)
      edge_mask_kernel<<<NEDGE / 256, 256, 0, stream>>>(eidx, topidx, emask);
  }

  gemm_end_kernel<<<(NNODE / 16) / 8, 256, 0, stream>>>(h, Wend, bend, topidx, out);
}
